// DeepFMmodule_37014028157165
// MI455X (gfx1250) — compile-verified, matrix operand order checked by reading, not run on hardware
//
#include <hip/hip_runtime.h>
#include <hip/hip_bf16.h>
#include <stdint.h>

// Problem constants (from reference)
#define Nn   16384
#define Ff   39
#define Vv   100000
#define Ee   16
#define H1n  400
#define H2n  400
#define FE   624          // F*E
#define K1p  640          // FE padded to multiple of 32
#define K2p  416          // H1 padded to multiple of 32

typedef __bf16 bf16_t;
typedef __attribute__((ext_vector_type(16))) __bf16 v16bf;
typedef __attribute__((ext_vector_type(8)))  float  v8f;

union FragU { v16bf v; uint4 q[2]; };

// A/B 16-bit fragment: per ISA layout, lane holds two contiguous 8-element
// (16-byte) K-chunks at offsets kb + 8*half and kb + 16 + 8*half.
__device__ __forceinline__ v16bf load_frag(const bf16_t* __restrict__ rowp, int kb, int half) {
  FragU f;
  f.q[0] = *reinterpret_cast<const uint4*>(rowp + kb + 8 * half);
  f.q[1] = *reinterpret_cast<const uint4*>(rowp + kb + 16 + 8 * half);
  return f.v;
}

__device__ __forceinline__ v8f wmma_bf16(v16bf a, v16bf b, v8f c) {
  // (neg_a, A, neg_b, B, c_mod, C, reuse_a, reuse_b)
  return __builtin_amdgcn_wmma_f32_16x16x32_bf16(false, a, false, b, (short)0, c, false, false);
}

__device__ __forceinline__ void split_store(float x, bf16_t* hi, bf16_t* lo) {
  bf16_t h = (bf16_t)x;
  *hi = h;
  *lo = (bf16_t)(x - (float)h);
}

// ---------------- Stage 0: column sums of Xv (for 'first' term) -------------
__global__ void dfm_colsum(const float* __restrict__ Xv, float* __restrict__ S) {
  __shared__ float red[256];
  int f = blockIdx.x;
  float acc = 0.f;
  for (int n = threadIdx.x; n < Nn; n += 256) acc += Xv[n * Ff + f];
  red[threadIdx.x] = acc;
  __syncthreads();
  for (int s = 128; s > 0; s >>= 1) {
    if ((int)threadIdx.x < s) red[threadIdx.x] += red[threadIdx.x + s];
    __syncthreads();
  }
  if (threadIdx.x == 0) S[f] = red[0];
}

// ---------------- Stage 1: weight split (fp32 -> bf16 hi/lo, transposed) ----
__global__ void dfm_split_w1(const float* __restrict__ W1,
                             bf16_t* __restrict__ hiT, bf16_t* __restrict__ loT) {
  int l = blockIdx.x * 256 + threadIdx.x;        // [0, H1n*K1p)
  int j = l / K1p, k = l % K1p;
  float w = (k < FE) ? W1[(size_t)k * H1n + j] : 0.f;
  split_store(w, &hiT[(size_t)j * K1p + k], &loT[(size_t)j * K1p + k]);
}

__global__ void dfm_split_w2(const float* __restrict__ W2,
                             bf16_t* __restrict__ hiT, bf16_t* __restrict__ loT) {
  int l = blockIdx.x * 256 + threadIdx.x;        // [0, H2n*K2p)
  int j = l / K2p, k = l % K2p;
  float w = (k < H1n) ? W2[(size_t)k * H2n + j] : 0.f;
  split_store(w, &hiT[(size_t)j * K2p + k], &loT[(size_t)j * K2p + k]);
}

__global__ void dfm_zero(float* __restrict__ p, int n) {
  int i = blockIdx.x * 256 + threadIdx.x;
  if (i < n) p[i] = 0.f;
}

// ---------------- Stage 2: gather + FM first/second + build X hi/lo ---------
__global__ void dfm_gather(const int* __restrict__ Xi, const float* __restrict__ Xv,
                           const float* __restrict__ emb1, const float* __restrict__ emb2,
                           const float* __restrict__ S,
                           bf16_t* __restrict__ Xhi, bf16_t* __restrict__ Xlo,
                           float* __restrict__ baseo) {
  int wave = threadIdx.x >> 5;
  int lane = threadIdx.x & 31;
  int n = blockIdx.x * 8 + wave;                 // one wave32 per sample
  int e = lane & 15;
  int fh = lane >> 4;

  float sacc = 0.f, ssacc = 0.f;
  for (int f = fh; f < Ff; f += 2) {
    int   id = Xi[n * Ff + f];
    float xv = Xv[n * Ff + f];
    float v  = emb2[((size_t)f * Vv + id) * Ee + e] * xv;
    sacc  += v;
    ssacc += v * v;
    size_t o = (size_t)n * K1p + f * Ee + e;
    split_store(v, &Xhi[o], &Xlo[o]);
  }
  if (lane < 16) {                                // zero the K padding 624..639
    size_t o = (size_t)n * K1p + FE + lane;
    Xhi[o] = (bf16_t)0.f;
    Xlo[o] = (bf16_t)0.f;
  }

  // second[n] = sum_e 0.5*(s_e^2 - ss_e); combine two f-halves, reduce over e
  float st  = sacc  + __shfl_xor(sacc,  16, 32);
  float sst = ssacc + __shfl_xor(ssacc, 16, 32);
  float sec = 0.5f * (st * st - sst);
  sec += __shfl_xor(sec, 8, 32);
  sec += __shfl_xor(sec, 4, 32);
  sec += __shfl_xor(sec, 2, 32);
  sec += __shfl_xor(sec, 1, 32);

  // first[n] = sum_f emb1[f, idx] * S[f]
  float acc1 = 0.f;
  for (int f = lane; f < Ff; f += 32) {
    int id = Xi[n * Ff + f];
    acc1 += emb1[(size_t)f * Vv + id] * S[f];
  }
  for (int m = 16; m > 0; m >>= 1) acc1 += __shfl_xor(acc1, m, 32);

  if (lane == 0) baseo[n] = acc1 + sec;
}

// ---------------- Stage 3: GEMM1 (X @ W1) with fused BN, split output -------
__global__ void __launch_bounds__(256)
dfm_gemm1(const bf16_t* __restrict__ Ahi, const bf16_t* __restrict__ Alo,
          const bf16_t* __restrict__ BhiT, const bf16_t* __restrict__ BloT,
          const float* __restrict__ b1, const float* __restrict__ g1,
          const float* __restrict__ beta1,
          bf16_t* __restrict__ Yhi, bf16_t* __restrict__ Ylo) {
  const float sc = rsqrtf(1.0f + 1e-5f);
  int wave = threadIdx.x >> 5, lane = threadIdx.x & 31;
  int tile = blockIdx.x * 8 + wave;
  int tm = tile / 26, tj = tile % 26;            // 26 tiles per tm; a block's 8 waves may span two tm
  int nl = lane & 15, half = lane >> 4;

  if (tj == 25) {                                // zero K2 padding cols 400..415
    #pragma unroll
    for (int r = 0; r < 8; ++r) {
      size_t o = (size_t)(tm * 16 + r + 8 * half) * K2p + H1n + nl;
      Yhi[o] = (bf16_t)0.f;
      Ylo[o] = (bf16_t)0.f;
    }
    return;
  }

  int row = tm * 16 + nl;
  int col = tj * 16 + nl;
  const bf16_t* ah = Ahi  + (size_t)row * K1p;
  const bf16_t* al = Alo  + (size_t)row * K1p;
  const bf16_t* bh = BhiT + (size_t)col * K1p;
  const bf16_t* bl = BloT + (size_t)col * K1p;

  v8f acc;
  #pragma unroll
  for (int i = 0; i < 8; ++i) acc[i] = 0.f;

  for (int kb = 0; kb < K1p; kb += 32) {
    if (kb + 32 < K1p) {
      __builtin_prefetch(ah + kb + 32, 0, 1);
      __builtin_prefetch(bh + kb + 32, 0, 1);
    }
    v16bf fah = load_frag(ah, kb, half);
    v16bf fbh = load_frag(bh, kb, half);
    v16bf fal = load_frag(al, kb, half);
    v16bf fbl = load_frag(bl, kb, half);
    acc = wmma_bf16(fah, fbh, acc);   // hi*hi
    acc = wmma_bf16(fah, fbl, acc);   // hi*lo
    acc = wmma_bf16(fal, fbh, acc);   // lo*hi  (compensated fp32-class product)
  }

  float bb = b1[col], gg = g1[col] * sc, bt = beta1[col];
  #pragma unroll
  for (int r = 0; r < 8; ++r) {
    float y = (acc[r] + bb) * gg + bt;
    size_t o = (size_t)(tm * 16 + r + 8 * half) * K2p + col;
    split_store(y, &Yhi[o], &Ylo[o]);
  }
}

// ---------------- Stage 4: GEMM2 (H1 @ W2) fused BN + row-sum ---------------
__global__ void __launch_bounds__(256)
dfm_gemm2(const bf16_t* __restrict__ Ahi, const bf16_t* __restrict__ Alo,
          const bf16_t* __restrict__ BhiT, const bf16_t* __restrict__ BloT,
          const float* __restrict__ b2, const float* __restrict__ g2,
          const float* __restrict__ beta2,
          float* __restrict__ hsum) {
  const float sc = rsqrtf(1.0f + 1e-5f);
  int wave = threadIdx.x >> 5, lane = threadIdx.x & 31;
  int tile = blockIdx.x * 8 + wave;
  int tm = tile / 25, tj = tile % 25;
  int nl = lane & 15, half = lane >> 4;
  int row = tm * 16 + nl;
  int col = tj * 16 + nl;

  const bf16_t* ah = Ahi  + (size_t)row * K2p;
  const bf16_t* al = Alo  + (size_t)row * K2p;
  const bf16_t* bh = BhiT + (size_t)col * K2p;
  const bf16_t* bl = BloT + (size_t)col * K2p;

  v8f acc;
  #pragma unroll
  for (int i = 0; i < 8; ++i) acc[i] = 0.f;

  for (int kb = 0; kb < K2p; kb += 32) {
    if (kb + 32 < K2p) {
      __builtin_prefetch(ah + kb + 32, 0, 1);
      __builtin_prefetch(bh + kb + 32, 0, 1);
    }
    v16bf fah = load_frag(ah, kb, half);
    v16bf fbh = load_frag(bh, kb, half);
    v16bf fal = load_frag(al, kb, half);
    v16bf fbl = load_frag(bl, kb, half);
    acc = wmma_bf16(fah, fbh, acc);
    acc = wmma_bf16(fah, fbl, acc);
    acc = wmma_bf16(fal, fbh, acc);
  }

  float bb = b2[col], gg = g2[col] * sc, bt = beta2[col];
  #pragma unroll
  for (int r = 0; r < 8; ++r) {
    float y = (acc[r] + bb) * gg + bt;     // 16 cols of row (tm*16 + r + 8*half)
    y += __shfl_xor(y, 8, 32);
    y += __shfl_xor(y, 4, 32);
    y += __shfl_xor(y, 2, 32);
    y += __shfl_xor(y, 1, 32);
    if (nl == 0) atomicAdd(&hsum[tm * 16 + r + 8 * half], y);
  }
}

// ---------------- Stage 5: finalize ----------------------------------------
__global__ void dfm_final(const float* __restrict__ baseo, const float* __restrict__ hsum,
                          const float* __restrict__ bias, float* __restrict__ out) {
  int n = blockIdx.x * 256 + threadIdx.x;
  out[n] = baseo[n] + hsum[n] + bias[0];
}

// ---------------------------------------------------------------------------
extern "C" void kernel_launch(void* const* d_in, const int* in_sizes, int n_in,
                              void* d_out, int out_size, void* d_ws, size_t ws_size,
                              hipStream_t stream) {
  (void)in_sizes; (void)n_in; (void)out_size; (void)ws_size;
  const int*   Xi    = (const int*)  d_in[0];
  const float* Xv    = (const float*)d_in[1];
  const float* emb1  = (const float*)d_in[2];
  const float* emb2  = (const float*)d_in[3];
  const float* W1    = (const float*)d_in[4];
  const float* b1    = (const float*)d_in[5];
  const float* W2    = (const float*)d_in[6];
  const float* b2    = (const float*)d_in[7];
  const float* g1    = (const float*)d_in[8];
  const float* beta1 = (const float*)d_in[9];
  const float* g2    = (const float*)d_in[10];
  const float* beta2 = (const float*)d_in[11];
  const float* bias  = (const float*)d_in[12];
  float* out = (float*)d_out;

  // Workspace carve-up (all chunks 256B-aligned)
  char* w = (char*)d_ws;
  float*  S     = (float*)w;  w += 256;
  float*  baseo = (float*)w;  w += (size_t)Nn * 4;
  float*  hsum  = (float*)w;  w += (size_t)Nn * 4;
  bf16_t* Xhi   = (bf16_t*)w; w += (size_t)Nn * K1p * 2;
  bf16_t* Xlo   = (bf16_t*)w; w += (size_t)Nn * K1p * 2;
  bf16_t* W1hiT = (bf16_t*)w; w += (size_t)H1n * K1p * 2;
  bf16_t* W1loT = (bf16_t*)w; w += (size_t)H1n * K1p * 2;
  bf16_t* Y1hi  = (bf16_t*)w; w += (size_t)Nn * K2p * 2;
  bf16_t* Y1lo  = (bf16_t*)w; w += (size_t)Nn * K2p * 2;
  bf16_t* W2hiT = (bf16_t*)w; w += (size_t)H2n * K2p * 2;
  bf16_t* W2loT = (bf16_t*)w; w += (size_t)H2n * K2p * 2;

  dfm_colsum <<<Ff, 256, 0, stream>>>(Xv, S);
  dfm_split_w1<<<(H1n * K1p) / 256, 256, 0, stream>>>(W1, W1hiT, W1loT);
  dfm_split_w2<<<(H2n * K2p) / 256, 256, 0, stream>>>(W2, W2hiT, W2loT);
  dfm_zero   <<<Nn / 256, 256, 0, stream>>>(hsum, Nn);
  dfm_gather <<<Nn / 8, 256, 0, stream>>>(Xi, Xv, emb1, emb2, S, Xhi, Xlo, baseo);
  dfm_gemm1  <<<(1024 * 26) / 8, 256, 0, stream>>>(Xhi, Xlo, W1hiT, W1loT,
                                                   b1, g1, beta1, Y1hi, Y1lo);
  dfm_gemm2  <<<(1024 * 25) / 8, 256, 0, stream>>>(Y1hi, Y1lo, W2hiT, W2loT,
                                                   b2, g2, beta2, hsum);
  dfm_final  <<<Nn / 256, 256, 0, stream>>>(baseo, hsum, bias, out);
}
